// Corm_68169720922792
// MI455X (gfx1250) — matrix operand order repaired and checked by reading, not compiled, not run on hardware
//
#include <hip/hip_runtime.h>

// Problem constants (match reference setup_inputs)
static constexpr int NH  = 32;    // heads
static constexpr int HD  = 128;   // head dim
static constexpr int NQ  = 2048;  // queries
static constexpr int NKV = 2048;  // keys

typedef __attribute__((ext_vector_type(16))) _Float16 v16h;
typedef __attribute__((ext_vector_type(8)))  _Float16 v8h;
typedef __attribute__((ext_vector_type(2)))  _Float16 v2h;
typedef __attribute__((ext_vector_type(8)))  float    v8f;

// packed f32x2 -> f16x2 conversion (v_cvt_pk_rtz_f16_f32)
__device__ __forceinline__ v2h pk2(float a, float b) {
  union { decltype(__builtin_amdgcn_cvt_pkrtz(0.f, 0.f)) p; v2h v; } u;
  u.p = __builtin_amdgcn_cvt_pkrtz(a, b);
  return u.v;
}

__device__ __forceinline__ float half_max16(float x) {
  x = fmaxf(x, __shfl_xor(x, 8, 32));
  x = fmaxf(x, __shfl_xor(x, 4, 32));
  x = fmaxf(x, __shfl_xor(x, 2, 32));
  x = fmaxf(x, __shfl_xor(x, 1, 32));
  return x;
}

// Load a WMMA A-operand chunk (f32 -> f16, pre-scaled). Row = base+(lane&15).
// lanes<16: K {0..15}; lanes>=16: K {16..31} of this 32-wide K chunk.
__device__ __forceinline__ v16h load_A_chunk(const float* rowptr, int c, int hi, float sc) {
  const float* b0 = rowptr + c * 32 + hi * 16;
  const float* b1 = rowptr + c * 32 + hi * 16 + 8;
  float4 f0 = *(const float4*)(b0);
  float4 f1 = *(const float4*)(b0 + 4);
  float4 f2 = *(const float4*)(b1);
  float4 f3 = *(const float4*)(b1 + 4);
  union { v16h v; v2h h[8]; } a;
  a.h[0] = pk2(f0.x * sc, f0.y * sc);
  a.h[1] = pk2(f0.z * sc, f0.w * sc);
  a.h[2] = pk2(f1.x * sc, f1.y * sc);
  a.h[3] = pk2(f1.z * sc, f1.w * sc);
  a.h[4] = pk2(f2.x * sc, f2.y * sc);
  a.h[5] = pk2(f2.z * sc, f2.w * sc);
  a.h[6] = pk2(f3.x * sc, f3.y * sc);
  a.h[7] = pk2(f3.z * sc, f3.w * sc);
  return a.v;
}

// ---------------------------------------------------------------------------
// One 32-key flash-attention chunk. MASKED=true applies the causal mask.
// Block-uniform: every wave of the block calls this for the same kv0 list.
// ---------------------------------------------------------------------------
template <bool MASKED>
__device__ __forceinline__ void fa_chunk(
    int kv0, int tid, int wave, int lh, int hi, int q0, int h, bool prefetch,
    const float* __restrict__ k, const float* __restrict__ v,
    _Float16 (*ks)[HD], _Float16 (*vt)[32], _Float16 (*ps)[16][32],
    const v16h (&Aq)[4], const v16h& onesB,
    v8f (&O)[8], v8f& Lacc, float (&m_run)[8])
{
  // ---- cooperative stage: K chunk (32 rows x 128) row-major f16 ----
  {
    const int row = tid >> 2, seg = tid & 3;
    const float* kp = k + ((size_t)(kv0 + row) * NH + h) * HD + seg * 32;
    _Float16* dst = &ks[row][seg * 32];
#pragma unroll
    for (int i = 0; i < 8; ++i) {
      float4 f = *(const float4*)(kp + i * 4);
      *(v2h*)&dst[i * 4]     = pk2(f.x, f.y);
      *(v2h*)&dst[i * 4 + 2] = pk2(f.z, f.w);
    }
    if (prefetch) __builtin_prefetch(kp + 32 * NH * HD, 0, 1);
  }
  // ---- cooperative stage: V chunk transposed -> vt[d][kv] f16 ----
  {
    const int pr = tid & 15;   // kv pair (rows 2pr, 2pr+1)
    const int dsg = tid >> 4;  // d segment of 16
    const float* va = v + ((size_t)(kv0 + 2 * pr) * NH + h) * HD + dsg * 16;
    const float* vb = va + (size_t)NH * HD;
#pragma unroll
    for (int i = 0; i < 16; ++i)
      *(v2h*)&vt[dsg * 16 + i][2 * pr] = pk2(va[i], vb[i]);
    if (prefetch) __builtin_prefetch(va + 32 * NH * HD, 0, 1);
  }
  __syncthreads();

  // ---- preload K B-operands, then S = Q K^T for two 16-key tiles ----
  v16h Bk0[4], Bk1[4];
#pragma unroll
  for (int c = 0; c < 4; ++c) {
    Bk0[c] = *(const v16h*)&ks[lh][c * 32 + hi * 16];
    Bk1[c] = *(const v16h*)&ks[16 + lh][c * 32 + hi * 16];
  }
  v8f S0, S1;
#pragma unroll
  for (int r = 0; r < 8; ++r) { S0[r] = 0.0f; S1[r] = 0.0f; }
#pragma unroll
  for (int c = 0; c < 4; ++c) {
    S0 = __builtin_amdgcn_wmma_f32_16x16x32_f16(false, Aq[c], false, Bk0[c],
                                                (short)0, S0, false, false);
    S1 = __builtin_amdgcn_wmma_f32_16x16x32_f16(false, Aq[c], false, Bk1[c],
                                                (short)0, S1, false, false);
  }

  // ---- online softmax (row max only; row sum comes from a WMMA) ----
  float alv[8];
  bool noscale = true;
#pragma unroll
  for (int r = 0; r < 8; ++r) {
    const int qrow = q0 + r + hi * 8;
    float s0 = S0[r], s1 = S1[r];              // already scaled via Q
    if (MASKED) {
      if (kv0 + lh      > qrow) s0 = -3.0e38f;
      if (kv0 + 16 + lh > qrow) s1 = -3.0e38f;
    }
    const float mt = half_max16(fmaxf(s0, s1));
    const float mn = fmaxf(m_run[r], mt);
    const float al = __expf(m_run[r] - mn);
    const float p0 = __expf(s0 - mn);
    const float p1 = __expf(s1 - mn);
    m_run[r] = mn;
    alv[r] = al;
    noscale = noscale && (al == 1.0f);
    ps[wave][r + hi * 8][lh]      = (_Float16)p0;
    ps[wave][r + hi * 8][16 + lh] = (_Float16)p1;
  }
  if (!__all(noscale)) {
#pragma unroll
    for (int r = 0; r < 8; ++r) {
#pragma unroll
      for (int dt = 0; dt < 8; ++dt) O[dt][r] *= alv[r];
      Lacc[r] *= alv[r];
    }
  }

  // ---- P back in A layout (same-wave LDS round trip, DS is in-order) ----
  v16h Ap;
  {
    v8h g0 = *(const v8h*)&ps[wave][lh][hi * 16];       // K {0..7}/{16..23}
    v8h g1 = *(const v8h*)&ps[wave][lh][hi * 16 + 8];  // K {8..15}/{24..31}
#pragma unroll
    for (int i = 0; i < 8; ++i) { Ap[i] = g0[i]; Ap[8 + i] = g1[i]; }
  }

  // ---- preload V B-operands, then O += P*V and Lacc += P*1 ----
  v16h Bv[8];
#pragma unroll
  for (int dt = 0; dt < 8; ++dt)
    Bv[dt] = *(const v16h*)&vt[dt * 16 + lh][hi * 16];
#pragma unroll
  for (int dt = 0; dt < 8; ++dt)
    O[dt] = __builtin_amdgcn_wmma_f32_16x16x32_f16(false, Ap, false, Bv[dt],
                                                   (short)0, O[dt], false, false);
  Lacc = __builtin_amdgcn_wmma_f32_16x16x32_f16(false, Ap, false, onesB,
                                                (short)0, Lacc, false, false);
  __syncthreads();
}

// ---------------------------------------------------------------------------
// Pass 1: flash attention forward. Block = 128 threads = 4 waves.
// Wave w handles q rows [q0b + 16w, q0b + 16w + 15] of head blockIdx.y.
// Writes per-row CORM threshold T[h][q] = m + log(l) + log(1/(q+1)) to ws.
// ---------------------------------------------------------------------------
__global__ __launch_bounds__(128) void corm_fwd_kernel(
    const float* __restrict__ q, const float* __restrict__ k,
    const float* __restrict__ v, const float* __restrict__ cm,
    float* __restrict__ out, float* __restrict__ Tws)
{
  __shared__ __align__(64) _Float16 ks[32][HD];      // K chunk, row-major f16   (8KB)
  __shared__ __align__(64) _Float16 vt[HD][32];      // V chunk, transposed f16  (8KB)
  __shared__ __align__(64) _Float16 ps[4][16][32];   // per-wave P staging       (4KB)

  const int tid  = threadIdx.x;
  const int wave = tid >> 5;
  const int lane = tid & 31;
  const int lh   = lane & 15;
  const int hi   = lane >> 4;
  const int h    = blockIdx.y;
  const int q0b  = blockIdx.x * 64;
  const int q0   = q0b + wave * 16;
  const float scale = 0.08838834764831845f; // 1/sqrt(128), folded into Q

  // Q tile in WMMA A layout (pre-scaled), 4 chunks of K=32 over D=128
  v16h Aq[4];
  {
    const float* qrow = q + ((size_t)(q0 + lh) * NH + h) * HD;
#pragma unroll
    for (int c = 0; c < 4; ++c) Aq[c] = load_A_chunk(qrow, c, hi, scale);
  }
  v16h onesB;
#pragma unroll
  for (int i = 0; i < 16; ++i) onesB[i] = (_Float16)1.0f;

  float m_run[8];
  v8f O[8], Lacc;
#pragma unroll
  for (int r = 0; r < 8; ++r) { m_run[r] = -3.0e38f; Lacc[r] = 0.0f; }
#pragma unroll
  for (int dt = 0; dt < 8; ++dt)
#pragma unroll
    for (int r = 0; r < 8; ++r) O[dt][r] = 0.0f;

  // Unmasked main chunks: keys [0, 64*bx) are strictly below every q row.
  const int nfull = blockIdx.x * 2;
  for (int kc = 0; kc < nfull; ++kc)
    fa_chunk<false>(kc << 5, tid, wave, lh, hi, q0, h, true,
                    k, v, ks, vt, ps, Aq, onesB, O, Lacc, m_run);
  // Two masked tail chunks covering the diagonal band [64*bx, 64*bx+64).
#pragma unroll
  for (int t = 0; t < 2; ++t)
    fa_chunk<true>((nfull + t) << 5, tid, wave, lh, hi, q0, h, t == 0,
                   k, v, ks, vt, ps, Aq, onesB, O, Lacc, m_run);

  // ---- epilogue: normalize, store out, store CORM thresholds ----
#pragma unroll
  for (int r = 0; r < 8; ++r) {
    const int qrow = q0 + r + hi * 8;
    const float l = Lacc[r];              // uniform across lanes (P x ones)
    const float inv = 1.0f / l;
#pragma unroll
    for (int dt = 0; dt < 8; ++dt)
      out[((size_t)qrow * NH + h) * HD + dt * 16 + lh] = O[dt][r] * inv;
    if (lh == 0) {
      const float thr = cm[(size_t)qrow * NKV];            // 1/(q+1)
      Tws[h * NQ + qrow] = m_run[r] + __logf(l) + __logf(thr);
    }
  }
}

// ---------------------------------------------------------------------------
// Pass 2: CORM per-key mask. Wave handles one 16-key tile of one head;
// iterates causal q-tiles, recomputing S with WMMA and OR-reducing s >= T_q.
// ---------------------------------------------------------------------------
__global__ __launch_bounds__(128) void corm_mask_kernel(
    const float* __restrict__ q, const float* __restrict__ k,
    const float* __restrict__ Tws, float* __restrict__ corm)
{
  const int tid  = threadIdx.x;
  const int wave = tid >> 5;
  const int lane = tid & 31;
  const int lh   = lane & 15;
  const int hi   = lane >> 4;
  const int h    = blockIdx.y;
  const int kvt  = blockIdx.x * 4 + wave;
  const int kv0  = kvt * 16;
  const float scale = 0.08838834764831845f;

  // B operand for this key tile: lane row kv0+lh, K(d) = {0..15}/{16..31} per chunk
  v16h Bk[4];
  {
    const float* kr = k + ((size_t)(kv0 + lh) * NH + h) * HD;
#pragma unroll
    for (int c = 0; c < 4; ++c) {
      const float* b = kr + c * 32 + hi * 16;
      float4 f0 = *(const float4*)(b);
      float4 f1 = *(const float4*)(b + 4);
      float4 f2 = *(const float4*)(b + 8);
      float4 f3 = *(const float4*)(b + 12);
      union { v16h v; v2h h[8]; } bb;
      bb.h[0] = pk2(f0.x, f0.y); bb.h[1] = pk2(f0.z, f0.w);
      bb.h[2] = pk2(f1.x, f1.y); bb.h[3] = pk2(f1.z, f1.w);
      bb.h[4] = pk2(f2.x, f2.y); bb.h[5] = pk2(f2.z, f2.w);
      bb.h[6] = pk2(f3.x, f3.y); bb.h[7] = pk2(f3.z, f3.w);
      Bk[c] = bb.v;
    }
  }

  float orv = 0.0f;
  for (int qt = kvt; qt < NQ / 16; ++qt) {
    const int q0 = qt * 16;
    const float* qrowp = q + ((size_t)(q0 + lh) * NH + h) * HD;
    v16h Aq[4];
#pragma unroll
    for (int c = 0; c < 4; ++c) Aq[c] = load_A_chunk(qrowp, c, hi, scale);

    v8f S;
#pragma unroll
    for (int r = 0; r < 8; ++r) S[r] = 0.0f;
#pragma unroll
    for (int c = 0; c < 4; ++c)
      S = __builtin_amdgcn_wmma_f32_16x16x32_f16(false, Aq[c], false, Bk[c],
                                                 (short)0, S, false, false);

#pragma unroll
    for (int r = 0; r < 8; ++r) {
      const int qrow = q0 + r + hi * 8;
      const float Tq = Tws[h * NQ + qrow];
      if ((kv0 + lh <= qrow) && (S[r] >= Tq)) orv = 1.0f;
    }
  }
  // combine the two half-waves (same key column, disjoint q-rows)
  orv = fmaxf(orv, __shfl_xor(orv, 16, 32));
  if (hi == 0) corm[(size_t)h * NKV + kv0 + lh] = orv;
}

extern "C" void kernel_launch(void* const* d_in, const int* in_sizes, int n_in,
                              void* d_out, int out_size, void* d_ws, size_t ws_size,
                              hipStream_t stream) {
  (void)in_sizes; (void)n_in; (void)out_size; (void)ws_size;
  const float* q  = (const float*)d_in[0];
  const float* k  = (const float*)d_in[1];
  const float* v  = (const float*)d_in[2];
  const float* cm = (const float*)d_in[3];
  float* out  = (float*)d_out;
  float* corm = out + (size_t)NQ * NH * HD;   // outputs concatenated: out, corm_score
  float* Tws  = (float*)d_ws;                 // NH*NQ floats of threshold scratch

  dim3 block(128);
  dim3 g1(NQ / 64, NH);
  corm_fwd_kernel<<<g1, block, 0, stream>>>(q, k, v, cm, out, Tws);
  dim3 g2(NKV / 16 / 4, NH);
  corm_mask_kernel<<<g2, block, 0, stream>>>(q, k, Tws, corm);
}
